// DecoderBlock_14766097564185
// MI455X (gfx1250) — matrix operand order repaired and checked by reading, not compiled, not run on hardware
//
#include <hip/hip_runtime.h>
#include <hip/hip_bf16.h>
#include <math.h>

typedef __attribute__((ext_vector_type(16))) _Float16 v16h;
typedef __attribute__((ext_vector_type(8)))  _Float16 v8h;
typedef __attribute__((ext_vector_type(8)))  float    v8f;
typedef __attribute__((ext_vector_type(4)))  int      v4i;

#define B_   8
#define S_   1024
#define D_   768
#define H_   12
#define DH_  64
#define D4_  (4 * D_)
#define N_   (B_ * S_)
#define EPS_ 1e-5f
#define LD_  72   // padded LDS row stride (halves); 144B -> 16B aligned, conflict-free

// ---------------------------------------------------------------------------
// Async global->LDS staging (CDNA5).
// ---------------------------------------------------------------------------
#if defined(__gfx1250__) && __has_builtin(__builtin_amdgcn_global_load_async_to_lds_b128)
#define USE_ASYNC 1
__device__ __forceinline__ void stage16(_Float16* lds, const _Float16* g) {
    __builtin_amdgcn_global_load_async_to_lds_b128(
        (__attribute__((address_space(1))) v4i*)g,
        (__attribute__((address_space(3))) v4i*)lds, 0, 0);
}
__device__ __forceinline__ void stage_fence() {
    asm volatile("s_wait_asynccnt 0x0" ::: "memory");
}
#else
#define USE_ASYNC 0
__device__ __forceinline__ void stage16(_Float16* lds, const _Float16* g) {
    *(v8h*)lds = *(const v8h*)g;
}
__device__ __forceinline__ void stage_fence() {}
#endif

// ---------------------------------------------------------------------------
// WMMA helpers (CDNA5 wave32, 16x16x32 f16 -> f32)
// ---------------------------------------------------------------------------
__device__ __forceinline__ v8f wmma16(v16h a, v16h b, v8f c) {
    return __builtin_amdgcn_wmma_f32_16x16x32_f16(
        false, a, false, b, (short)0, c, false, false);
}

// A fragment 16x32: src row-major [m][k], stride ld halves (16B-aligned rows).
__device__ __forceinline__ v16h load_Afrag(const _Float16* src, int ld) {
    int lane = threadIdx.x & 31;
    const _Float16* p = src + (lane & 15) * ld + (lane >> 4) * 8;
    v8h lo = *(const v8h*)p;
    v8h hi = *(const v8h*)(p + 16);
    v16h a;
#pragma unroll
    for (int e = 0; e < 8; ++e) { a[e] = lo[e]; a[e + 8] = hi[e]; }
    return a;
}

// B fragment 32x16 from an N-major (transposed) tile: srcT[n][k], stride ld.
__device__ __forceinline__ v16h load_BTfrag(const _Float16* srcT, int ld) {
    int lane = threadIdx.x & 31;
    const _Float16* p = srcT + (lane & 15) * ld + (lane >> 4) * 8;
    v8h lo = *(const v8h*)p;
    v8h hi = *(const v8h*)(p + 16);
    v16h b;
#pragma unroll
    for (int e = 0; e < 8; ++e) { b[e] = lo[e]; b[e + 8] = hi[e]; }
    return b;
}

__device__ __forceinline__ float gelu_exact(float x) {
    return 0.5f * x * (1.0f + erff(x * 0.70710678118654752f));
}

// ---------------------------------------------------------------------------
// LayerNorm: one block (256 threads) per row of 768, emits f16
// ---------------------------------------------------------------------------
__global__ void ln_kernel(const float* __restrict__ x,
                          const float* __restrict__ gamma,
                          const float* __restrict__ beta,
                          _Float16* __restrict__ y) {
    __shared__ float red[256];
    int row = blockIdx.x;
    int tid = threadIdx.x;
    const float* xr = x + (size_t)row * D_;
    float v0 = xr[tid], v1 = xr[tid + 256], v2 = xr[tid + 512];
    red[tid] = v0 + v1 + v2;
    __syncthreads();
#pragma unroll
    for (int off = 128; off > 0; off >>= 1) {
        if (tid < off) red[tid] += red[tid + off];
        __syncthreads();
    }
    float mean = red[0] * (1.0f / D_);
    __syncthreads();
    float d0 = v0 - mean, d1 = v1 - mean, d2 = v2 - mean;
    red[tid] = d0 * d0 + d1 * d1 + d2 * d2;
    __syncthreads();
#pragma unroll
    for (int off = 128; off > 0; off >>= 1) {
        if (tid < off) red[tid] += red[tid + off];
        __syncthreads();
    }
    float rstd = rsqrtf(red[0] * (1.0f / D_) + EPS_);
    _Float16* yr = y + (size_t)row * D_;
    yr[tid]       = (_Float16)(d0 * rstd * gamma[tid]       + beta[tid]);
    yr[tid + 256] = (_Float16)(d1 * rstd * gamma[tid + 256] + beta[tid + 256]);
    yr[tid + 512] = (_Float16)(d2 * rstd * gamma[tid + 512] + beta[tid + 512]);
}

// ---------------------------------------------------------------------------
// FFN weights fp32 -> f16
// ---------------------------------------------------------------------------
__global__ void cvtw_kernel(const float* __restrict__ W1, const float* __restrict__ W2,
                            _Float16* __restrict__ w1h, _Float16* __restrict__ w2h) {
    int i = blockIdx.x * 256 + threadIdx.x;
    if (i < D_ * D4_) {
        w1h[i] = (_Float16)W1[i];
        w2h[i] = (_Float16)W2[i];
    }
}

// ---------------------------------------------------------------------------
// QKV projection: block 128 (4 waves), grid (S/64, B*H).
// ---------------------------------------------------------------------------
__global__ void qkv_kernel(const _Float16* __restrict__ xn,
                           const float* __restrict__ Wq, const float* __restrict__ bq,
                           const float* __restrict__ Wk, const float* __restrict__ bk,
                           const float* __restrict__ Wv, const float* __restrict__ bv,
                           _Float16* __restrict__ qh, _Float16* __restrict__ kh,
                           _Float16* __restrict__ vh) {
    __shared__ alignas(16) _Float16 sX[64 * LD_];
    __shared__ alignas(16) _Float16 sWT[3][64 * LD_];
    __shared__ float sBias[3][64];

    int tid = threadIdx.x;
    int bh = blockIdx.y;
    int b = bh / H_, h = bh % H_;
    int s0 = blockIdx.x * 64;
    size_t nrow0 = (size_t)b * S_ + s0;

    for (int j = tid; j < 64 * 8; j += 128) {
        int r = j >> 3, c8 = (j & 7) * 8;
        stage16(sX + r * LD_ + c8, xn + (nrow0 + r) * D_ + h * DH_ + c8);
    }
    const float* wsrc[3] = { Wq + h * DH_ * DH_, Wk + h * DH_ * DH_, Wv + h * DH_ * DH_ };
#pragma unroll
    for (int mat = 0; mat < 3; ++mat)
        for (int i = tid; i < 64 * 64; i += 128) {
            int d = i >> 6, e = i & 63;
            sWT[mat][e * LD_ + d] = (_Float16)wsrc[mat][i];
        }
    if (tid < 64) {
        sBias[0][tid] = bq[h * DH_ + tid];
        sBias[1][tid] = bk[h * DH_ + tid];
        sBias[2][tid] = bv[h * DH_ + tid];
    }
    stage_fence();
    __syncthreads();

    int wave = tid >> 5, lane = tid & 31;
    int n = lane & 15, g = lane >> 4;
    int mrow = wave * 16;
    size_t base = ((size_t)bh * S_ + s0) * DH_;

#pragma unroll
    for (int mat = 0; mat < 3; ++mat) {
        v8f acc[4];
#pragma unroll
        for (int t = 0; t < 4; ++t) {
            float bias = sBias[mat][t * 16 + n];
#pragma unroll
            for (int r = 0; r < 8; ++r) acc[t][r] = bias;
        }
#pragma unroll
        for (int kk = 0; kk < 64; kk += 32) {
            v16h a = load_Afrag(sX + mrow * LD_ + kk, LD_);
#pragma unroll
            for (int t = 0; t < 4; ++t) {
                v16h bb = load_BTfrag(&sWT[mat][(t * 16) * LD_ + kk], LD_);
                acc[t] = wmma16(a, bb, acc[t]);
            }
        }
        _Float16* dst = (mat == 0) ? qh : (mat == 1) ? kh : vh;
        float scale = (mat == 0) ? 0.125f : 1.0f;
#pragma unroll
        for (int t = 0; t < 4; ++t)
#pragma unroll
            for (int r = 0; r < 8; ++r)
                dst[base + (size_t)(mrow + r + 8 * g) * DH_ + t * 16 + n] =
                    (_Float16)(acc[t][r] * scale);
    }
}

// ---------------------------------------------------------------------------
// Flash attention per (b,h): block 128 (4 waves), 64 query rows.
// K/V tiles double-buffered: next tile's async staging overlaps this tile's
// WMMAs. One barrier per iteration.
// ---------------------------------------------------------------------------
__global__ void attn_kernel(const float* __restrict__ x,
                            const _Float16* __restrict__ qh,
                            const _Float16* __restrict__ kh,
                            const _Float16* __restrict__ vh,
                            float* __restrict__ out) {
    __shared__ alignas(16) _Float16 sQ[64 * LD_];
    __shared__ alignas(16) _Float16 sK[2][64 * LD_];
    __shared__ alignas(16) _Float16 sVT[2][64 * LD_];
    __shared__ alignas(16) _Float16 sP[4][16 * LD_];

    int tid = threadIdx.x, wave = tid >> 5, lane = tid & 31;
    int n = lane & 15, g = lane >> 4;
    int bh = blockIdx.y;
    int b = bh / H_, h = bh % H_;
    int s0 = blockIdx.x * 64;
    size_t hbase = (size_t)bh * S_ * DH_;

    auto stageKV = [&](int buf, int j) {
        for (int i = tid; i < 64 * 8; i += 128) {
            int r = i >> 3, c8 = (i & 7) * 8;
            stage16(&sK[buf][r * LD_ + c8], kh + hbase + (size_t)(j * 64 + r) * DH_ + c8);
        }
        for (int i = tid; i < 64 * 8; i += 128) {
            int r = i >> 3, c8 = (i & 7) * 8;
            v8h v = *(const v8h*)(vh + hbase + (size_t)(j * 64 + r) * DH_ + c8);
#pragma unroll
            for (int e = 0; e < 8; ++e) sVT[buf][(c8 + e) * LD_ + r] = v[e];
        }
    };

    for (int j = tid; j < 64 * 8; j += 128) {
        int r = j >> 3, c8 = (j & 7) * 8;
        stage16(sQ + r * LD_ + c8, qh + hbase + (size_t)(s0 + r) * DH_ + c8);
    }
    stageKV(0, 0);
    stage_fence();
    __syncthreads();

    v8f o[4] = {};
    float mrowv[8], lrowv[8];
#pragma unroll
    for (int r = 0; r < 8; ++r) { mrowv[r] = -1e30f; lrowv[r] = 0.0f; }
    int mrow = wave * 16;
    int cur = 0;

    for (int j = 0; j < S_ / 64; ++j) {
        int nxt = cur ^ 1;
        if (j + 1 < S_ / 64) stageKV(nxt, j + 1);  // overlaps this tile's math

        // S = Q @ K^T
        v8f sc[4] = {};
#pragma unroll
        for (int kk = 0; kk < 64; kk += 32) {
            v16h a = load_Afrag(sQ + mrow * LD_ + kk, LD_);
#pragma unroll
            for (int t = 0; t < 4; ++t) {
                v16h bb = load_BTfrag(&sK[cur][(t * 16) * LD_ + kk], LD_);
                sc[t] = wmma16(a, bb, sc[t]);
            }
        }

        // Online softmax (row = r + 8*g, replicated across 16 lanes)
#pragma unroll
        for (int r = 0; r < 8; ++r) {
            float v = sc[0][r];
#pragma unroll
            for (int t = 1; t < 4; ++t) v = fmaxf(v, sc[t][r]);
#pragma unroll
            for (int mask = 1; mask < 16; mask <<= 1)
                v = fmaxf(v, __shfl_xor(v, mask, 32));
            float mnew = fmaxf(mrowv[r], v);
            float alpha = __expf(mrowv[r] - mnew);
            mrowv[r] = mnew;
            float rs = 0.0f;
#pragma unroll
            for (int t = 0; t < 4; ++t) {
                float p = __expf(sc[t][r] - mnew);
                sc[t][r] = p;
                rs += p;
            }
#pragma unroll
            for (int mask = 1; mask < 16; mask <<= 1)
                rs += __shfl_xor(rs, mask, 32);
            lrowv[r] = lrowv[r] * alpha + rs;
#pragma unroll
            for (int t = 0; t < 4; ++t) o[t][r] *= alpha;
        }

        // P staged through wave-private LDS (in-wave DS ordering, no barrier)
#pragma unroll
        for (int t = 0; t < 4; ++t)
#pragma unroll
            for (int r = 0; r < 8; ++r)
                sP[wave][(r + 8 * g) * LD_ + t * 16 + n] = (_Float16)sc[t][r];

        // O += P @ V
#pragma unroll
        for (int kk = 0; kk < 64; kk += 32) {
            v16h a = load_Afrag(&sP[wave][kk], LD_);
#pragma unroll
            for (int t = 0; t < 4; ++t) {
                v16h bb = load_BTfrag(&sVT[cur][(t * 16) * LD_ + kk], LD_);
                o[t] = wmma16(a, bb, o[t]);
            }
        }

        stage_fence();     // next buffer's async transfers complete
        __syncthreads();   // all waves done reading cur before it is reused
        cur = nxt;
    }

    // Epilogue: out = x + O / l
#pragma unroll
    for (int t = 0; t < 4; ++t)
#pragma unroll
        for (int r = 0; r < 8; ++r) {
            int row = s0 + mrow + r + 8 * g;
            int col = h * DH_ + t * 16 + n;
            size_t gi = ((size_t)b * S_ + row) * D_ + col;
            out[gi] = x[gi] + o[t][r] / lrowv[r];
        }
}

// ---------------------------------------------------------------------------
// 128x64-tile WMMA GEMM, double-buffered: C = A(MxK,f16) @ Bw(KxN,f16).
// Each wave owns 32 rows -> every B-fragment feeds 2 WMMAs.
// EPI 0: C16 = f16(gelu(acc));  EPI 1: Cf += acc.  block 128 (4 waves)
// ---------------------------------------------------------------------------
template <int EPI>
__global__ void gemm128_kernel(const _Float16* __restrict__ A,
                               const _Float16* __restrict__ Bw,
                               _Float16* __restrict__ C16,
                               float* __restrict__ Cf,
                               int M, int K, int N) {
    __shared__ alignas(16) _Float16 sA[2][128 * LD_];
    __shared__ alignas(16) _Float16 sBT[2][64 * LD_];
    int tid = threadIdx.x, wave = tid >> 5, lane = tid & 31;
    int n = lane & 15, g = lane >> 4;
    int m0 = blockIdx.y * 128, n0 = blockIdx.x * 64;
    int mrow = wave * 32;

    auto stageTiles = [&](int buf, int kt) {
        for (int j = tid; j < 128 * 8; j += 128) {
            int r = j >> 3, c8 = (j & 7) * 8;
            stage16(&sA[buf][r * LD_ + c8], A + (size_t)(m0 + r) * K + kt + c8);
        }
        for (int j = tid; j < 64 * 8; j += 128) {
            int r = j >> 3, c8 = (j & 7) * 8;
            v8h v = *(const v8h*)(Bw + (size_t)(kt + r) * N + n0 + c8);
#pragma unroll
            for (int e = 0; e < 8; ++e) sBT[buf][(c8 + e) * LD_ + r] = v[e];
        }
    };

    v8f acc[2][4] = {};
    stageTiles(0, 0);
    stage_fence();
    __syncthreads();

    int cur = 0;
    for (int kt = 0; kt < K; kt += 64) {
        int nxt = cur ^ 1;
        if (kt + 64 < K) stageTiles(nxt, kt + 64);  // overlaps WMMAs below
#pragma unroll
        for (int kk = 0; kk < 64; kk += 32) {
            v16h a0 = load_Afrag(&sA[cur][(mrow +  0) * LD_ + kk], LD_);
            v16h a1 = load_Afrag(&sA[cur][(mrow + 16) * LD_ + kk], LD_);
#pragma unroll
            for (int t = 0; t < 4; ++t) {
                v16h bb = load_BTfrag(&sBT[cur][(t * 16) * LD_ + kk], LD_);
                acc[0][t] = wmma16(a0, bb, acc[0][t]);
                acc[1][t] = wmma16(a1, bb, acc[1][t]);
            }
        }
        stage_fence();
        __syncthreads();
        cur = nxt;
    }

#pragma unroll
    for (int u = 0; u < 2; ++u)
#pragma unroll
        for (int t = 0; t < 4; ++t)
#pragma unroll
            for (int r = 0; r < 8; ++r) {
                size_t row = (size_t)(m0 + mrow + u * 16 + r + 8 * g);
                size_t col = (size_t)(n0 + t * 16 + n);
                if (EPI == 0) {
                    C16[row * N + col] = (_Float16)gelu_exact(acc[u][t][r]);
                } else {
                    Cf[row * N + col] += acc[u][t][r];
                }
            }
}

// ---------------------------------------------------------------------------
// Launch
// ---------------------------------------------------------------------------
extern "C" void kernel_launch(void* const* d_in, const int* in_sizes, int n_in,
                              void* d_out, int out_size, void* d_ws, size_t ws_size,
                              hipStream_t stream) {
    const float* x     = (const float*)d_in[0];
    const float* ln1_g = (const float*)d_in[1];
    const float* ln1_b = (const float*)d_in[2];
    const float* Wq    = (const float*)d_in[3];
    const float* bq    = (const float*)d_in[4];
    const float* Wk    = (const float*)d_in[5];
    const float* bk    = (const float*)d_in[6];
    const float* Wv    = (const float*)d_in[7];
    const float* bv    = (const float*)d_in[8];
    const float* ln2_g = (const float*)d_in[9];
    const float* ln2_b = (const float*)d_in[10];
    const float* W1    = (const float*)d_in[11];
    const float* W2    = (const float*)d_in[12];
    float* out = (float*)d_out;

    _Float16* ws  = (_Float16*)d_ws;
    _Float16* xn  = ws;
    _Float16* qh  = xn  + (size_t)N_ * D_;
    _Float16* kh  = qh  + (size_t)N_ * D_;
    _Float16* vh  = kh  + (size_t)N_ * D_;
    _Float16* h2  = vh  + (size_t)N_ * D_;
    _Float16* w1h = h2  + (size_t)N_ * D_;
    _Float16* w2h = w1h + (size_t)D_ * D4_;
    _Float16* act = w2h + (size_t)D4_ * D_;

    cvtw_kernel<<<(D_ * D4_ + 255) / 256, 256, 0, stream>>>(W1, W2, w1h, w2h);
    ln_kernel<<<N_, 256, 0, stream>>>(x, ln1_g, ln1_b, xn);
    qkv_kernel<<<dim3(S_ / 64, B_ * H_), 128, 0, stream>>>(
        xn, Wq, bq, Wk, bk, Wv, bv, qh, kh, vh);
    attn_kernel<<<dim3(S_ / 64, B_ * H_), 128, 0, stream>>>(x, qh, kh, vh, out);
    ln_kernel<<<N_, 256, 0, stream>>>(out, ln2_g, ln2_b, h2);
    gemm128_kernel<0><<<dim3(D4_ / 64, N_ / 128), 128, 0, stream>>>(
        h2, w1h, act, nullptr, N_, D_, D4_);
    gemm128_kernel<1><<<dim3(D_ / 64, N_ / 128), 128, 0, stream>>>(
        act, w2h, nullptr, out, N_, D4_, D_);
}
